// MajConv_66314295050711
// MI455X (gfx1250) — 8-bit matrix operand order repaired and checked by reading, compiled, not run on hardware
//
#include <hip/hip_runtime.h>

typedef __attribute__((ext_vector_type(8))) int v8i;
typedef unsigned int u32;

#define NB   16
#define CI   64
#define CO   64
#define HH   32
#define WW   32
#define NPIX (NB * CI * HH * WW)

__global__ __launch_bounds__(256)
void binarize_triples(const float* __restrict__ x, u32* __restrict__ T) {
  const int idx = blockIdx.x * 256 + threadIdx.x;
  const int px  = idx & (WW - 1);
  const float vm = (px > 0)      ? x[idx - 1] : 0.f;
  const float v0 =                 x[idx];
  const float vp = (px < WW - 1) ? x[idx + 1] : 0.f;
  const u32 sm = (u32)(((vm > 0.f) - (vm < 0.f)) & 0xff);
  const u32 s0 = (u32)(((v0 > 0.f) - (v0 < 0.f)) & 0xff);
  const u32 sp = (u32)(((vp > 0.f) - (vp < 0.f)) & 0xff);
  T[idx] = sm | (s0 << 8) | (sp << 16);
}

__global__ __launch_bounds__(256)
void majconv_wmma_iu8(const u32* __restrict__ T,
                      const float* __restrict__ w,
                      float* __restrict__ out) {
  __shared__ int4 wsm[CI * 16];

  const int tid  = threadIdx.x;
  const int lane = tid & 31;
  const int wave = tid >> 5;
  const bool lolane = (lane < 16);
  const int nidx = lane & 15;

  int blk = blockIdx.x;
  const int b   = blk >> 5;  blk &= 31;
  const int cot = blk >> 3;
  const int ptg = blk & 7;
  const int co0 = cot * 16;

  for (int idx = tid; idx < CI * 16; idx += 256) {
    const int ci = idx >> 4;
    const int co = co0 + (idx & 15);
    unsigned p0 = 0, p1 = 0, p2 = 0;
#pragma unroll
    for (int t = 0; t < 9; ++t) {
      const float v = w[(co * 9 + t) * CI + ci];
      const unsigned u = (unsigned)(((v > 0.f) - (v < 0.f)) & 0xff);
      if (t < 4)      p0 |= u << (8 * t);
      else if (t < 8) p1 |= u << (8 * (t - 4));
      else            p2 |= u;
    }
    wsm[idx] = make_int4((int)p0, (int)p1, (int)p2, 0);
  }
  __syncthreads();

  const int pt   = ptg * 8 + wave;
  const int py   = pt >> 1;
  const int px0  = (pt & 1) * 16;
  const int px   = px0 + nidx;
  const int base = py * WW + px;

  int acc[8];
#pragma unroll
  for (int j = 0; j < 8; ++j) acc[j] = 0;

  for (int ci = 0; ci < CI; ++ci) {
    const u32* Tb = T + ((b * CI + ci) * (HH * WW));
    if (ci + 1 < CI)
      __builtin_prefetch(Tb + HH * WW + base, 0, 3);

    u32 Tm1 = 0, Tp1 = 0;
    if (py > 0)      Tm1 = Tb[base - WW];
    const u32 T0 =         Tb[base];
    if (py < HH - 1) Tp1 = Tb[base + WW];

    const u32 a0lo = Tm1 | (T0 << 24);
    const u32 a1lo = (T0 >> 8) | (Tp1 << 16);
    const u32 a0hi = (Tp1 >> 16) & 0xff;
    v8i A = {};
    A[0] = lolane ? (int)a0lo : (int)a0hi;
    A[1] = lolane ? (int)a1lo : 0;

    const int4 q = wsm[ci * 16 + nidx];
    v8i Bm = {};
    Bm[0] = lolane ? q.x : q.z;
    Bm[1] = lolane ? q.y : 0;
    Bm[2] = 0;

    v8i Cz = {};
    v8i D = __builtin_amdgcn_wmma_i32_16x16x64_iu8(true, A, true, Bm, Cz,
                                                   false, false);

#pragma unroll
    for (int j = 0; j < 8; ++j) {
      int v = D[j];
      v = v > 1 ? 1 : v;
      v = v < -1 ? -1 : v;
      acc[j] += v;
    }
  }

  const int co    = co0 + nidx;
  const int mbase = lolane ? 0 : 8;
  float* o = out + ((b * CO + co) * (HH * WW)) + py * WW + px0 + mbase;
  const float4 r0 = make_float4((float)acc[0], (float)acc[1],
                                (float)acc[2], (float)acc[3]);
  const float4 r1 = make_float4((float)acc[4], (float)acc[5],
                                (float)acc[6], (float)acc[7]);
  *(float4*)(o)     = r0;
  *(float4*)(o + 4) = r1;
}

extern "C" void kernel_launch(void* const* d_in, const int* in_sizes, int n_in,
                              void* d_out, int out_size, void* d_ws, size_t ws_size,
                              hipStream_t stream) {
  (void)in_sizes; (void)n_in; (void)out_size; (void)ws_size;
  const float* x = (const float*)d_in[0];
  const float* w = (const float*)d_in[1];
  float* out = (float*)d_out;
  u32* T = (u32*)d_ws;

  hipLaunchKernelGGL(binarize_triples, dim3(NPIX / 256), dim3(256), 0, stream,
                     x, T);
  hipLaunchKernelGGL(majconv_wmma_iu8, dim3(NB * 4 * 8), dim3(256), 0, stream,
                     T, w, out);
}
